// ParallelRetention_33363305955513
// MI455X (gfx1250) — compile-verified, matrix operand order checked by reading, not run on hardware
//
#include <hip/hip_runtime.h>
#include <hip/hip_bf16.h>

// ---------------------------------------------------------------------------
// ParallelRetention fused kernels for gfx1250 (MI455X).
// bf16 WMMA + f32 accumulate; all fragments are 2 x ds_load_b128;
// tile movement uses GLOBAL_LOAD_ASYNC_TO_LDS_B128 with double buffering
// (ASYNCcnt-tracked DMA overlapped with WMMA compute).
// ---------------------------------------------------------------------------

#define T_SZ   4096
#define D_SZ   256
#define B_SZ   8

typedef __attribute__((ext_vector_type(16))) __bf16 v16bf;
typedef __attribute__((ext_vector_type(8)))  __bf16 v8bf;
typedef __attribute__((ext_vector_type(8)))  float  v8f;

// LDS row strides (elements); row stride in bytes is 16B-divisible.
#define LDB  264   // bf16 264*2 = 528 B
#define LDV  72    // bf16 Vt / score tiles: 72*2 = 144 B
#define LDG  68    // f32 gamma tile: 68*4 = 272 B

// LDS byte offsets for retention_kernel
#define OFF_Q    0
#define OFF_R    33792
#define OFF_S    67584
#define OFF_K    76800      // + buf * 33792
#define OFF_VT   144384     // + buf * 36864
#define OFF_G    218112     // + buf * 17408
#define OFF_WO   76800      // phase-2 overlay

// ---- gfx1250 async global->LDS DMA (ASYNCcnt) ------------------------------
__device__ __forceinline__ void async_b128(void* lds_dst, const void* sbase,
                                           int voff_bytes) {
    unsigned ldsa = (unsigned)(unsigned long long)(uintptr_t)lds_dst;
    asm volatile("global_load_async_to_lds_b128 %0, %1, %2"
                 :: "v"(ldsa), "v"(voff_bytes), "s"(sbase)
                 : "memory");
}
__device__ __forceinline__ void wait_async_20() {   // allow next tile in flight
    asm volatile("s_wait_asynccnt 0x14" ::: "memory");
}
__device__ __forceinline__ void wait_async_0() {
    asm volatile("s_wait_asynccnt 0x0" ::: "memory");
}

// Fragment loader: lane holds row `row`; elems 0..7 = kbase+koff+0..7,
// elems 8..15 = kbase+16+koff+0..7 (koff=8 for lanes 16..31), per ISA 7.12.2.
__device__ __forceinline__ v16bf frag_rowK(const __bf16* __restrict__ base,
                                           int row, int stride, int kbase, int koff) {
    const __bf16* p = base + row * stride + kbase + koff;
    v8bf lo = *(const v8bf*)(p);
    v8bf hi = *(const v8bf*)(p + 16);
    return __builtin_shufflevector(lo, hi, 0, 1, 2, 3, 4, 5, 6, 7,
                                   8, 9, 10, 11, 12, 13, 14, 15);
}

__device__ __forceinline__ v8f wmma_bf16(v16bf a, v16bf b, v8f c) {
    return __builtin_amdgcn_wmma_f32_16x16x32_bf16(false, a, false, b, (short)0, c,
                                                   false, false);
}

// ---------------------------------------------------------------------------
// Kernel 0: W -> bf16 W^T (one-time).  grid = (16 tiles, 4 matrices), block 256.
// ---------------------------------------------------------------------------
__global__ __launch_bounds__(256) void wt_transpose_kernel(
    const float* __restrict__ Wq, const float* __restrict__ Wk,
    const float* __restrict__ Wv, const float* __restrict__ Wo,
    __bf16* __restrict__ wqt, __bf16* __restrict__ wkt,
    __bf16* __restrict__ wvt, __bf16* __restrict__ wot)
{
    __shared__ float tlds[64][68];
    const float* W; __bf16* WT;
    if (blockIdx.y == 0)      { W = Wq; WT = wqt; }
    else if (blockIdx.y == 1) { W = Wk; WT = wkt; }
    else if (blockIdx.y == 2) { W = Wv; WT = wvt; }
    else                      { W = Wo; WT = wot; }

    const int kt = (blockIdx.x >> 2) * 64;
    const int nt = (blockIdx.x & 3) * 64;
    const int tid = threadIdx.x;

#pragma unroll
    for (int i = 0; i < 4; ++i) {
        int chunk = tid + i * 256;
        int row = chunk >> 4, c = chunk & 15;
        float4 w = *(const float4*)(W + (size_t)(kt + row) * D_SZ + nt + c * 4);
        tlds[row][c * 4 + 0] = w.x; tlds[row][c * 4 + 1] = w.y;
        tlds[row][c * 4 + 2] = w.z; tlds[row][c * 4 + 3] = w.w;
    }
    __syncthreads();
#pragma unroll
    for (int i = 0; i < 2; ++i) {
        int chunk = tid + i * 256;
        int n = chunk >> 3, c = chunk & 7;
        alignas(16) __bf16 h[8];
#pragma unroll
        for (int j = 0; j < 8; ++j) h[j] = (__bf16)tlds[c * 8 + j][n];
        *(uint4*)(WT + (size_t)(nt + n) * D_SZ + kt + c * 8) = *(const uint4*)h;
    }
}

// ---------------------------------------------------------------------------
// Kernel 1: q/k/v projection.  grid = (T/64, B, 3); block = 256 (8 waves).
// W^T tile arrives via async DMA; x converts f32->bf16 through VGPRs.
// LDS: wtlds 256 x LDB @0 (135168), xlds 64 x LDB @135168 (33792) = 168960 B.
// ---------------------------------------------------------------------------
__global__ __launch_bounds__(256) void qkv_proj_kernel(
    const float* __restrict__ x,
    const __bf16* __restrict__ wqt, const __bf16* __restrict__ wkt,
    const __bf16* __restrict__ wvt,
    const float* __restrict__ bq, const float* __restrict__ bk,
    const float* __restrict__ bv,
    __bf16* __restrict__ qg, __bf16* __restrict__ kg, __bf16* __restrict__ vtg)
{
    extern __shared__ char smem[];
    __bf16* wtlds = (__bf16*)smem;              // 256(n) x LDB(k)
    __bf16* xlds  = (__bf16*)(smem + 135168);   // 64(m) x LDB(k)

    const int b  = blockIdx.y;
    const int m0 = blockIdx.x * 64;
    const int tid  = threadIdx.x;
    const int lane = tid & 31;
    const int wave = tid >> 5;
    const int hi   = (lane >> 4) & 1;
    const int lc   = lane & 15;
    const int koff = hi * 8;
    const int mt   = wave >> 1;
    const int nh   = wave & 1;

    const __bf16* WT; const float* bias;
    if (blockIdx.z == 0)      { WT = wqt; bias = bq; }
    else if (blockIdx.z == 1) { WT = wkt; bias = bk; }
    else                      { WT = wvt; bias = bv; }

    // W^T -> LDS via async DMA: 8192 uint4 chunks / 256 thr = 32 each
#pragma unroll
    for (int i = 0; i < 32; ++i) {
        int chunk = tid + i * 256;
        int row = chunk >> 5, c = chunk & 31;
        async_b128(wtlds + row * LDB + c * 8, WT, row * 512 + c * 16);
    }
    // x tile (f32 -> bf16), packed 8B stores
    const float* xb = x + (size_t)b * T_SZ * D_SZ;
#pragma unroll 4
    for (int i = 0; i < 16; ++i) {
        int chunk = tid + i * 256;
        int row = chunk >> 6, c = chunk & 63;
        float4 w = *(const float4*)(xb + (size_t)(m0 + row) * D_SZ + c * 4);
        alignas(8) __bf16 h[4] = {(__bf16)w.x, (__bf16)w.y, (__bf16)w.z, (__bf16)w.w};
        *(uint2*)(xlds + row * LDB + c * 4) = *(const uint2*)h;
    }
    wait_async_0();
    __syncthreads();

    v8f acc[8];
#pragma unroll
    for (int i = 0; i < 8; ++i) acc[i] = v8f{};

#pragma unroll
    for (int kb = 0; kb < D_SZ; kb += 32) {
        v16bf A = frag_rowK(xlds, mt * 16 + lc, LDB, kb, koff);
#pragma unroll
        for (int ni = 0; ni < 8; ++ni) {
            v16bf Bf = frag_rowK(wtlds, nh * 128 + ni * 16 + lc, LDB, kb, koff);
            acc[ni] = wmma_bf16(A, Bf, acc[ni]);
        }
    }

    if (blockIdx.z < 2) {
        // repack through (now dead) xlds so global stores are coalesced uint4
        __bf16* dst = (blockIdx.z == 0) ? qg : kg;
        __bf16* db = dst + (size_t)b * T_SZ * D_SZ;
        __syncthreads();
#pragma unroll
        for (int ni = 0; ni < 8; ++ni) {
            int col = nh * 128 + ni * 16 + lc;
            float bb = bias[col];
#pragma unroll
            for (int r = 0; r < 8; ++r)
                xlds[(mt * 16 + r + hi * 8) * LDB + col] = (__bf16)(acc[ni][r] + bb);
        }
        __syncthreads();
#pragma unroll
        for (int i = 0; i < 8; ++i) {
            int chunk = tid + i * 256;
            int row = chunk >> 5, c = chunk & 31;
            *(uint4*)(db + (size_t)(m0 + row) * D_SZ + c * 8) =
                *(const uint4*)(xlds + row * LDB + c * 8);
        }
    } else {
        // v stored transposed [d][s]: lane's 8 rows are contiguous -> uint4
        __bf16* vb = vtg + (size_t)b * D_SZ * T_SZ;
#pragma unroll
        for (int ni = 0; ni < 8; ++ni) {
            int col = nh * 128 + ni * 16 + lc;
            float bb = bias[col];
            alignas(16) __bf16 h[8];
#pragma unroll
            for (int r = 0; r < 8; ++r) h[r] = (__bf16)(acc[ni][r] + bb);
            *(uint4*)(vb + (size_t)col * T_SZ + m0 + mt * 16 + hi * 8) =
                *(const uint4*)h;
        }
    }
}

// ---------------------------------------------------------------------------
// Kernel 2: fused retention.  grid = (T/64, B); block = 256 (8 waves).
// Double-buffered async DMA of K / V^T / gamma tiles overlapped with WMMA.
// LDS layout (bytes):
//   qlds   @0       64 x LDB bf16   (33792)
//   rlds   @33792   64 x LDB bf16   (33792)
//   slds   @67584   4 x 16 x LDV bf16 (9216)
//   kbuf   @76800   2 x 33792
//   vtbuf  @144384  2 x 36864
//   gbuf   @218112  2 x 17408      (end 252928)
//   wlds   @76800   256 x LDB bf16 (135168)  Wo^T overlay, phase 2
// total = 252928 B (< 320 KB WGP LDS; 1 WG / WGP by design)
// ---------------------------------------------------------------------------
__global__ __launch_bounds__(256) void retention_kernel(
    const __bf16* __restrict__ qg, const __bf16* __restrict__ kg,
    const __bf16* __restrict__ vtg, const __bf16* __restrict__ wot,
    const float* __restrict__ gamma, const float* __restrict__ bo,
    const float* __restrict__ prelu, float* __restrict__ out)
{
    extern __shared__ char smem[];
    __bf16* qlds = (__bf16*)(smem + OFF_Q);
    __bf16* rlds = (__bf16*)(smem + OFF_R);
    __bf16* slds = (__bf16*)(smem + OFF_S);
    __bf16* wlds = (__bf16*)(smem + OFF_WO);   // phase-2 overlay

    const int b  = blockIdx.y;
    const int m0 = blockIdx.x * 64;
    const int tid  = threadIdx.x;
    const int lane = tid & 31;
    const int wave = tid >> 5;
    const int hi   = (lane >> 4) & 1;
    const int lc   = lane & 15;
    const int koff = hi * 8;
    const int mt   = wave >> 1;   // 16-row tile of the 64-row block
    const int nh   = wave & 1;    // 128-col half of D_int

    const __bf16* qb = qg + (size_t)b * T_SZ * D_SZ;
    const __bf16* kb = kg + (size_t)b * T_SZ * D_SZ;
    const __bf16* vb = vtg + (size_t)b * D_SZ * T_SZ;   // [d][s]

    // issue tile DMA: 8 (K) + 8 (V^T) + 4 (gamma) = 20 async ops per thread
    auto issue_tile = [&](int sb, int buf) {
        __bf16* kd  = (__bf16*)(smem + OFF_K  + buf * 33792);
        __bf16* vtd = (__bf16*)(smem + OFF_VT + buf * 36864);
        float*  gd  = (float*) (smem + OFF_G  + buf * 17408);
#pragma unroll
        for (int i = 0; i < 8; ++i) {
            int chunk = tid + i * 256;
            int row = chunk >> 5, c = chunk & 31;
            async_b128(kd + row * LDB + c * 8, kb, (sb + row) * 512 + c * 16);
        }
#pragma unroll
        for (int i = 0; i < 8; ++i) {
            int chunk = tid + i * 256;
            int row = chunk >> 3, c = chunk & 7;
            async_b128(vtd + row * LDV + c * 8, vb,
                       row * (T_SZ * 2) + (sb + c * 8) * 2);
        }
#pragma unroll
        for (int i = 0; i < 4; ++i) {
            int chunk = tid + i * 256;
            int row = chunk >> 4, c = chunk & 15;
            async_b128(gd + row * LDG + c * 4, gamma,
                       ((m0 + row) * T_SZ + sb + c * 4) * 4);
        }
    };

    // q tile DMA (8 ops), then tile 0
#pragma unroll
    for (int i = 0; i < 8; ++i) {
        int chunk = tid + i * 256;
        int row = chunk >> 5, c = chunk & 31;
        async_b128(qlds + row * LDB + c * 8, qb, (m0 + row) * 512 + c * 16);
    }
    issue_tile(0, 0);

    v8f acc[8];
#pragma unroll
    for (int i = 0; i < 8; ++i) acc[i] = v8f{};

    // wait for q (tile0's 20 may stay in flight), preload invariant A-frags
    wait_async_20();
    __syncthreads();
    v16bf qfr[8];
#pragma unroll
    for (int kk = 0; kk < 8; ++kk)
        qfr[kk] = frag_rowK(qlds, mt * 16 + lc, LDB, kk * 32, koff);

    for (int it = 0; it < T_SZ / 64; ++it) {
        const int sb  = it * 64;
        const int cur = it & 1;
        const bool more = (it + 1 < T_SZ / 64);

        if (more) { issue_tile(sb + 64, cur ^ 1); wait_async_20(); }
        else      { wait_async_0(); }
        __syncthreads();                       // tile `it` visible to all

        const __bf16* kl  = (const __bf16*)(smem + OFF_K  + cur * 33792);
        const __bf16* vtl = (const __bf16*)(smem + OFF_VT + cur * 36864);
        const float*  gl  = (const float*) (smem + OFF_G  + cur * 17408);

        // ---- scores: wave computes si = nh*2 .. nh*2+1 for m-tile mt ------
        __bf16* sm = slds + mt * (16 * LDV);
#pragma unroll
        for (int s2 = 0; s2 < 2; ++s2) {
            int si = nh * 2 + s2;
            v8f c8 = v8f{};
#pragma unroll
            for (int kk = 0; kk < 8; ++kk) {
                v16bf Bf = frag_rowK(kl, si * 16 + lc, LDB, kk * 32, koff); // k^T
                c8 = wmma_bf16(qfr[kk], Bf, c8);
            }
#pragma unroll
            for (int r = 0; r < 8; ++r) {
                int lrow = r + hi * 8;
                float g = gl[(mt * 16 + lrow) * LDG + si * 16 + lc];
                sm[lrow * LDV + si * 16 + lc] = (__bf16)(c8[r] * g);
            }
        }
        __syncthreads();

        // ---- retained += gated_scores[16x64] @ V[64 x 128-half] -----------
#pragma unroll
        for (int kk = 0; kk < 64; kk += 32) {
            v16bf A = frag_rowK(sm, lc, LDV, kk, koff);
#pragma unroll
            for (int ni = 0; ni < 8; ++ni) {
                v16bf Bf = frag_rowK(vtl, nh * 128 + ni * 16 + lc, LDV, kk, koff);
                acc[ni] = wmma_bf16(A, Bf, acc[ni]);
            }
        }
        __syncthreads();   // tile fully consumed; next issue may overwrite
    }

    // ---- phase 2: y = retained @ Wo + bo, PReLU ---------------------------
#pragma unroll
    for (int ni = 0; ni < 8; ++ni)
#pragma unroll
        for (int r = 0; r < 8; ++r)
            rlds[(mt * 16 + r + hi * 8) * LDB + nh * 128 + ni * 16 + lc] =
                (__bf16)acc[ni][r];

    // Wo^T -> overlay LDS via async DMA
#pragma unroll
    for (int i = 0; i < 32; ++i) {
        int chunk = tid + i * 256;
        int row = chunk >> 5, c = chunk & 31;
        async_b128(wlds + row * LDB + c * 8, wot, row * 512 + c * 16);
    }
    wait_async_0();
    __syncthreads();

    v8f y[8];
#pragma unroll
    for (int i = 0; i < 8; ++i) y[i] = v8f{};
#pragma unroll
    for (int kk = 0; kk < D_SZ; kk += 32) {
        v16bf A = frag_rowK(rlds, mt * 16 + lc, LDB, kk, koff);
#pragma unroll
        for (int ni = 0; ni < 8; ++ni) {
            v16bf Bf = frag_rowK(wlds, nh * 128 + ni * 16 + lc, LDB, kk, koff);
            y[ni] = wmma_bf16(A, Bf, y[ni]);
        }
    }

    const float pa = prelu[0];
    float* ob = out + (size_t)b * T_SZ * D_SZ;
#pragma unroll
    for (int ni = 0; ni < 8; ++ni) {
        int col = nh * 128 + ni * 16 + lc;
        float bb = bo[col];
#pragma unroll
        for (int r = 0; r < 8; ++r) {
            int row = m0 + mt * 16 + r + hi * 8;
            float v = y[ni][r] + bb;
            v = (v >= 0.0f) ? v : pa * v;
            ob[(size_t)row * D_SZ + col] = v;
        }
    }
}

// ---------------------------------------------------------------------------
// Host launcher
// ---------------------------------------------------------------------------
extern "C" void kernel_launch(void* const* d_in, const int* in_sizes, int n_in,
                              void* d_out, int out_size, void* d_ws, size_t ws_size,
                              hipStream_t stream) {
    const float* x     = (const float*)d_in[0];
    const float* gamma = (const float*)d_in[1];
    const float* Wq    = (const float*)d_in[2];
    const float* bq    = (const float*)d_in[3];
    const float* Wk    = (const float*)d_in[4];
    const float* bk    = (const float*)d_in[5];
    const float* Wv    = (const float*)d_in[6];
    const float* bv    = (const float*)d_in[7];
    const float* Wo    = (const float*)d_in[8];
    const float* bo    = (const float*)d_in[9];
    const float* pre   = (const float*)d_in[10];
    float* out = (float*)d_out;

    const size_t tensor_bytes = (size_t)B_SZ * T_SZ * D_SZ * sizeof(__bf16); // 16 MB
    const size_t wt_bytes = (size_t)D_SZ * D_SZ * sizeof(__bf16);            // 128 KB
    char* ws = (char*)d_ws;
    __bf16* qg  = (__bf16*)(ws);
    __bf16* kg  = (__bf16*)(ws + tensor_bytes);
    __bf16* vtg = (__bf16*)(ws + 2 * tensor_bytes);
    __bf16* wqt = (__bf16*)(ws + 3 * tensor_bytes);
    __bf16* wkt = (__bf16*)(ws + 3 * tensor_bytes + wt_bytes);
    __bf16* wvt = (__bf16*)(ws + 3 * tensor_bytes + 2 * wt_bytes);
    __bf16* wot = (__bf16*)(ws + 3 * tensor_bytes + 3 * wt_bytes);

    wt_transpose_kernel<<<dim3(16, 4), 256, 0, stream>>>(Wq, Wk, Wv, Wo,
                                                         wqt, wkt, wvt, wot);

    dim3 g1(T_SZ / 64, B_SZ, 3);
    qkv_proj_kernel<<<g1, 256, 168960, stream>>>(x, wqt, wkt, wvt,
                                                 bq, bk, bv, qg, kg, vtg);

    dim3 g2(T_SZ / 64, B_SZ);
    retention_kernel<<<g2, 256, 252928, stream>>>(qg, kg, vtg, wot, gamma,
                                                  bo, pre, out);
}
